// MockLMHead_4733053960281
// MI455X (gfx1250) — matrix operand order repaired and checked by reading, not compiled, not run on hardware
//
#include <hip/hip_runtime.h>
#include <hip/hip_bf16.h>

// ---------------------------------------------------------------------------
// MockLMHead: logits = hidden @ W^T  (f32 in, f32 accum via bf16 WMMA),
// loss = mean over valid tokens of (logsumexp(row) - row[label]).
// d_out = [loss, logits...]
//
// Fast path: one-shot f32->bf16 convert of hidden & W into d_ws, then a
// bf16-fed WMMA GEMM using GLOBAL_LOAD_ASYNC_TO_LDS_B128 with triple-buffered
// LDS (no VGPR staging -> no spills, 1 barrier/K-step). Fallbacks kept for
// toolchains without the async builtin / small workspace.
// ---------------------------------------------------------------------------

typedef __attribute__((ext_vector_type(16))) __bf16 v16bf;
typedef __attribute__((ext_vector_type(8)))  __bf16 v8bf;
typedef __attribute__((ext_vector_type(8)))  float  v8f;
typedef __attribute__((__vector_size__(16))) int    vi128;   // async builtin's pointee

#define HH 2048        // hidden dim (K)
#define VV 32000       // vocab (N of GEMM)
#define BM 128         // token tile
#define BK 32          // K tile (matches wmma 16x16x32 K)
#define LDK 40         // padded LDS K-stride in bf16 elems (80 B, 16B aligned)
#define IGNORE_INDEX (-100)

#define AS1 __attribute__((address_space(1)))
#define AS3 __attribute__((address_space(3)))

#if __has_builtin(__builtin_amdgcn_global_load_async_to_lds_b128)
#define HAVE_ASYNC_LDS 1
#endif

#if __has_builtin(__builtin_amdgcn_s_wait_asynccnt)
#define WAIT_ASYNC(n) __builtin_amdgcn_s_wait_asynccnt(n)
#else
#define WAIT_ASYNC(n) asm volatile("s_wait_asynccnt %0" :: "i"(n) : "memory")
#endif

#ifdef HAVE_ASYNC_LDS
#define BNF 256        // vocab tile of fast kernel (async path)
#else
#define BNF 128        // vocab tile of fast kernel (reg-staged path)
#endif

__device__ __forceinline__ unsigned int pack_bf16x2(float a, float b) {
  unsigned int ua = __float_as_uint(a);
  unsigned int ub = __float_as_uint(b);
  ua = (ua + 0x7FFFu + ((ua >> 16) & 1u)) >> 16;   // round-to-nearest-even
  ub = (ub + 0x7FFFu + ((ub >> 16) & 1u)) >> 16;
  return ua | (ub << 16);
}

#ifdef HAVE_ASYNC_LDS
__device__ __forceinline__ void async_ld_b128(const void* g, void* l) {
  __builtin_amdgcn_global_load_async_to_lds_b128(
      (AS1 vi128*)g, (AS3 vi128*)l, 0, 0);
}
#endif

// ---------------------- one-shot f32 -> bf16 convert -----------------------
__global__ __launch_bounds__(256)
void cvt_f32_bf16_kernel(const float* __restrict__ src,
                         unsigned short* __restrict__ dst,
                         long long n4)              // count of float4 chunks
{
  long long i = (long long)blockIdx.x * 256 + threadIdx.x;
  long long stride = (long long)gridDim.x * 256;
  for (; i < n4; i += stride) {
    float4 v = ((const float4*)src)[i];
    uint2 p;
    p.x = pack_bf16x2(v.x, v.y);
    p.y = pack_bf16x2(v.z, v.w);
    ((uint2*)dst)[i] = p;
  }
}

#ifdef HAVE_ASYNC_LDS
// ------------------- bf16 GEMM, async global->LDS (fast path) ---------------
// Block: 128 tokens x 256 vocab, 8 waves as 2(M) x 4(N); wave tile 64x64.
// Triple-buffered LDS; per wave per K-step: 6 async b128 issues, 16 wmma.
__global__ __launch_bounds__(256, 1)
void lmhead_gemm_bf16_kernel(const unsigned short* __restrict__ hiddenb,
                             const unsigned short* __restrict__ Wb,
                             float* __restrict__ logits)
{
  __shared__ __align__(16) unsigned short ldsA[3][BM  * LDK];   // 3 x 10 KB
  __shared__ __align__(16) unsigned short ldsB[3][BNF * LDK];   // 3 x 20 KB

  const int tid  = threadIdx.x;
  const int lane = tid & 31;
  const int wid  = tid >> 5;
  const int wm   = wid & 1;        // 0..1 -> 64 token rows
  const int wn   = wid >> 1;       // 0..3 -> 64 vocab cols
  const int half = lane >> 4;
  const int l15  = lane & 15;

  const int m0 = blockIdx.x * BM;
  const int n0 = blockIdx.y * BNF;

  // A tile: 128x32 bf16 = 512 x 16B chunks -> 2/thread
  // B tile: 256x32 bf16 = 1024 x 16B chunks -> 4/thread
  int aRow[2], aKc[2], bRow[4], bKc[4];
#pragma unroll
  for (int p = 0; p < 2; ++p) { int idx = tid + p * 256; aRow[p] = idx >> 2; aKc[p] = idx & 3; }
#pragma unroll
  for (int p = 0; p < 4; ++p) { int idx = tid + p * 256; bRow[p] = idx >> 2; bKc[p] = idx & 3; }

  auto issueTile = [&](int k0, int buf) {
#pragma unroll
    for (int p = 0; p < 2; ++p) {
      const unsigned short* g = hiddenb + (size_t)(m0 + aRow[p]) * HH + k0 + aKc[p] * 8;
      async_ld_b128(g, &ldsA[buf][aRow[p] * LDK + aKc[p] * 8]);
    }
#pragma unroll
    for (int p = 0; p < 4; ++p) {
      const unsigned short* g = Wb + (size_t)(n0 + bRow[p]) * HH + k0 + bKc[p] * 8;
      async_ld_b128(g, &ldsB[buf][bRow[p] * LDK + bKc[p] * 8]);
    }
  };

  v8f acc[4][4] = {};
  const int nk = HH / BK;          // 64 K-steps

  issueTile(0, 0);
  issueTile(BK, 1);

  for (int ks = 0; ks < nk; ++ks) {
    // tile ks landed when our own first group of 6 async loads completed
    if (ks + 1 < nk) WAIT_ASYNC(6); else WAIT_ASYNC(0);
    __syncthreads();               // all waves' chunks of tile ks visible
    if (ks + 2 < nk) issueTile((ks + 2) * BK, (ks + 2) % 3);

    const int cur = ks % 3;
    // B frags: lane n=l15, K = half*8 + {0..7}, 16 + half*8 + {0..7}
    v16bf bF[4];
#pragma unroll
    for (int j = 0; j < 4; ++j) {
      const unsigned short* base = &ldsB[cur][(wn * 64 + j * 16 + l15) * LDK + half * 8];
      v8bf lo = *(const v8bf*)(const void*)(base);
      v8bf hi = *(const v8bf*)(const void*)(base + 16);
      bF[j] = __builtin_shufflevector(lo, hi, 0,1,2,3,4,5,6,7,8,9,10,11,12,13,14,15);
    }
    // A frags loaded one sub-tile at a time to limit register pressure
#pragma unroll
    for (int i = 0; i < 4; ++i) {
      const unsigned short* base = &ldsA[cur][(wm * 64 + i * 16 + l15) * LDK + half * 8];
      v8bf lo = *(const v8bf*)(const void*)(base);
      v8bf hi = *(const v8bf*)(const void*)(base + 16);
      v16bf aF = __builtin_shufflevector(lo, hi, 0,1,2,3,4,5,6,7,8,9,10,11,12,13,14,15);
#pragma unroll
      for (int j = 0; j < 4; ++j)
        acc[i][j] = __builtin_amdgcn_wmma_f32_16x16x32_bf16(
            false, aF, false, bF[j], (short)0, acc[i][j], false, false);
    }
  }

#pragma unroll
  for (int i = 0; i < 4; ++i)
#pragma unroll
    for (int j = 0; j < 4; ++j)
#pragma unroll
      for (int r = 0; r < 8; ++r) {
        int gm = m0 + wm * 64 + i * 16 + half * 8 + r;
        int gn = n0 + wn * 64 + j * 16 + l15;
        logits[(size_t)gm * VV + gn] = acc[i][j][r];
      }
}

#else  // !HAVE_ASYNC_LDS
// ------------- bf16 GEMM, register-staged (spill-free shape) ---------------
// Block: 128x128, 8 waves as 2(M) x 4(N); wave tile 64x32.
__global__ __launch_bounds__(256, 1)
void lmhead_gemm_bf16_kernel(const unsigned short* __restrict__ hiddenb,
                             const unsigned short* __restrict__ Wb,
                             float* __restrict__ logits)
{
  __shared__ __align__(16) unsigned short ldsA[2][BM  * LDK];
  __shared__ __align__(16) unsigned short ldsB[2][BNF * LDK];

  const int tid  = threadIdx.x;
  const int lane = tid & 31;
  const int wid  = tid >> 5;
  const int wm   = wid >> 2;       // 0..1
  const int wn   = wid & 3;        // 0..3
  const int half = lane >> 4;
  const int l15  = lane & 15;

  const int m0 = blockIdx.x * BM;
  const int n0 = blockIdx.y * BNF;

  int aRow[2], aKc[2];
#pragma unroll
  for (int p = 0; p < 2; ++p) { int idx = tid + p * 256; aRow[p] = idx >> 2; aKc[p] = idx & 3; }

  uint4 stA[2], stB[2];
  auto loadStage = [&](int k0) {
#pragma unroll
    for (int p = 0; p < 2; ++p) {
      stA[p] = *(const uint4*)(const void*)(hiddenb + (size_t)(m0 + aRow[p]) * HH + k0 + aKc[p] * 8);
      stB[p] = *(const uint4*)(const void*)(Wb      + (size_t)(n0 + aRow[p]) * HH + k0 + aKc[p] * 8);
    }
  };
  auto storeLds = [&](int buf) {
#pragma unroll
    for (int p = 0; p < 2; ++p) {
      *(uint4*)(void*)&ldsA[buf][aRow[p] * LDK + aKc[p] * 8] = stA[p];
      *(uint4*)(void*)&ldsB[buf][aRow[p] * LDK + aKc[p] * 8] = stB[p];
    }
  };

  v8f acc[4][2] = {};
  const int nk = HH / BK;
  loadStage(0);
  storeLds(0);
  loadStage(BK);
  __syncthreads();
  for (int ks = 0; ks < nk; ++ks) {
    const int cur = ks & 1;
    if (ks + 1 < nk) storeLds(cur ^ 1);
    if (ks + 2 < nk) loadStage((ks + 2) * BK);

    v16bf bF[2];
#pragma unroll
    for (int j = 0; j < 2; ++j) {
      const unsigned short* base = &ldsB[cur][(wn * 32 + j * 16 + l15) * LDK + half * 8];
      v8bf lo = *(const v8bf*)(const void*)(base);
      v8bf hi = *(const v8bf*)(const void*)(base + 16);
      bF[j] = __builtin_shufflevector(lo, hi, 0,1,2,3,4,5,6,7,8,9,10,11,12,13,14,15);
    }
#pragma unroll
    for (int i = 0; i < 4; ++i) {
      const unsigned short* base = &ldsA[cur][(wm * 64 + i * 16 + l15) * LDK + half * 8];
      v8bf lo = *(const v8bf*)(const void*)(base);
      v8bf hi = *(const v8bf*)(const void*)(base + 16);
      v16bf aF = __builtin_shufflevector(lo, hi, 0,1,2,3,4,5,6,7,8,9,10,11,12,13,14,15);
#pragma unroll
      for (int j = 0; j < 2; ++j)
        acc[i][j] = __builtin_amdgcn_wmma_f32_16x16x32_bf16(
            false, aF, false, bF[j], (short)0, acc[i][j], false, false);
    }
    __syncthreads();
  }
#pragma unroll
  for (int i = 0; i < 4; ++i)
#pragma unroll
    for (int j = 0; j < 2; ++j)
#pragma unroll
      for (int r = 0; r < 8; ++r) {
        int gm = m0 + wm * 64 + i * 16 + half * 8 + r;
        int gn = n0 + wn * 32 + j * 16 + l15;
        logits[(size_t)gm * VV + gn] = acc[i][j][r];
      }
}
#endif // HAVE_ASYNC_LDS

// --------------- inline-converting GEMM (small-ws fallback) ----------------
__global__ __launch_bounds__(256, 1)
void lmhead_gemm_f32_kernel(const float* __restrict__ hidden,
                            const float* __restrict__ W,
                            float* __restrict__ logits)
{
  __shared__ __align__(16) unsigned short ldsA[BM * LDK];
  __shared__ __align__(16) unsigned short ldsB[128 * LDK];

  const int tid  = threadIdx.x;
  const int lane = tid & 31;
  const int wid  = tid >> 5;
  const int wm   = wid >> 2;
  const int wn   = wid & 3;
  const int half = lane >> 4;
  const int l15  = lane & 15;

  const int m0 = blockIdx.x * BM;
  const int n0 = blockIdx.y * 128;

  int rowIdx[4], kqIdx[4];
#pragma unroll
  for (int p = 0; p < 4; ++p) {
    int idx = tid + p * 256;
    rowIdx[p] = idx >> 3;
    kqIdx[p]  = idx & 7;
  }

  float4 stA[4], stB[4];
  auto loadStage = [&](int k0) {
#pragma unroll
    for (int p = 0; p < 4; ++p) {
      stA[p] = *(const float4*)(const void*)(hidden + (size_t)(m0 + rowIdx[p]) * HH + k0 + kqIdx[p] * 4);
      stB[p] = *(const float4*)(const void*)(W      + (size_t)(n0 + rowIdx[p]) * HH + k0 + kqIdx[p] * 4);
    }
  };

  v8f acc[4][2] = {};
  loadStage(0);
  const int nk = HH / BK;
  for (int ks = 0; ks < nk; ++ks) {
    __syncthreads();
#pragma unroll
    for (int p = 0; p < 4; ++p) {
      uint2 pa, pb;
      pa.x = pack_bf16x2(stA[p].x, stA[p].y);
      pa.y = pack_bf16x2(stA[p].z, stA[p].w);
      pb.x = pack_bf16x2(stB[p].x, stB[p].y);
      pb.y = pack_bf16x2(stB[p].z, stB[p].w);
      *(uint2*)(void*)&ldsA[rowIdx[p] * LDK + kqIdx[p] * 4] = pa;
      *(uint2*)(void*)&ldsB[rowIdx[p] * LDK + kqIdx[p] * 4] = pb;
    }
    __syncthreads();
    if (ks + 1 < nk) loadStage((ks + 1) * BK);

    v16bf bF[2];
#pragma unroll
    for (int j = 0; j < 2; ++j) {
      const unsigned short* base = &ldsB[(wn * 32 + j * 16 + l15) * LDK + half * 8];
      v8bf lo = *(const v8bf*)(const void*)(base);
      v8bf hi = *(const v8bf*)(const void*)(base + 16);
      bF[j] = __builtin_shufflevector(lo, hi, 0,1,2,3,4,5,6,7,8,9,10,11,12,13,14,15);
    }
#pragma unroll
    for (int i = 0; i < 4; ++i) {
      const unsigned short* base = &ldsA[(wm * 64 + i * 16 + l15) * LDK + half * 8];
      v8bf lo = *(const v8bf*)(const void*)(base);
      v8bf hi = *(const v8bf*)(const void*)(base + 16);
      v16bf aF = __builtin_shufflevector(lo, hi, 0,1,2,3,4,5,6,7,8,9,10,11,12,13,14,15);
#pragma unroll
      for (int j = 0; j < 2; ++j)
        acc[i][j] = __builtin_amdgcn_wmma_f32_16x16x32_bf16(
            false, aF, false, bF[j], (short)0, acc[i][j], false, false);
    }
  }
#pragma unroll
  for (int i = 0; i < 4; ++i)
#pragma unroll
    for (int j = 0; j < 2; ++j)
#pragma unroll
      for (int r = 0; r < 8; ++r) {
        int gm = m0 + wm * 64 + i * 16 + half * 8 + r;
        int gn = n0 + wn * 32 + j * 16 + l15;
        logits[(size_t)gm * VV + gn] = acc[i][j][r];
      }
}

// --------------------------- per-row NLL -----------------------------------
__global__ __launch_bounds__(256)
void lmhead_loss_rows_kernel(const float* __restrict__ logits,
                             const long long* __restrict__ labels,
                             float* __restrict__ ws, int n_rows)
{
  __shared__ float sm[256];
  __shared__ float ss[256];
  const int r   = blockIdx.x;
  const int tid = threadIdx.x;
  const float* row = logits + (size_t)r * VV;

  float m = -3.4e38f, s = 0.f;
  for (int v = tid; v < VV; v += 256) {
    float x  = row[v];
    float nm = fmaxf(m, x);
    s = s * __expf(m - nm) + __expf(x - nm);
    m = nm;
  }
  sm[tid] = m; ss[tid] = s;
  __syncthreads();
  for (int off = 128; off > 0; off >>= 1) {
    if (tid < off) {
      float m1 = sm[tid], s1 = ss[tid];
      float m2 = sm[tid + off], s2 = ss[tid + off];
      float nm = fmaxf(m1, m2);
      sm[tid] = nm;
      ss[tid] = s1 * __expf(m1 - nm) + s2 * __expf(m2 - nm);
    }
    __syncthreads();
  }
  if (tid == 0) {
    long long lab = labels[r];
    bool valid = (lab != IGNORE_INDEX);
    float nll = 0.f;
    if (valid) nll = sm[0] + __logf(ss[0]) - row[(int)lab];
    ws[r]          = nll;
    ws[n_rows + r] = valid ? 1.f : 0.f;
  }
}

// ------------------------------ finalize -----------------------------------
__global__ __launch_bounds__(256)
void lmhead_finalize_kernel(const float* __restrict__ ws, int n_rows,
                            float* __restrict__ out)
{
  __shared__ float s_sum[256];
  __shared__ float s_cnt[256];
  const int tid = threadIdx.x;
  float sum = 0.f, cnt = 0.f;
  for (int i = tid; i < n_rows; i += 256) {   // fixed order -> deterministic
    sum += ws[i];
    cnt += ws[n_rows + i];
  }
  s_sum[tid] = sum; s_cnt[tid] = cnt;
  __syncthreads();
  for (int off = 128; off > 0; off >>= 1) {
    if (tid < off) {
      s_sum[tid] += s_sum[tid + off];
      s_cnt[tid] += s_cnt[tid + off];
    }
    __syncthreads();
  }
  if (tid == 0) out[0] = s_sum[0] / fmaxf(s_cnt[0], 1.f);
}

// ------------------------------ launcher -----------------------------------
extern "C" void kernel_launch(void* const* d_in, const int* in_sizes, int n_in,
                              void* d_out, int out_size, void* d_ws, size_t ws_size,
                              hipStream_t stream)
{
  const float*     hidden = (const float*)d_in[0];       // [B,S,H] f32
  const float*     W      = (const float*)d_in[1];       // [V,H]   f32
  const long long* labels = (const long long*)d_in[2];   // [B,S]   i64

  float* out    = (float*)d_out;
  float* logits = out + 1;                 // outputs: [loss, logits...]

  const int Ntok = in_sizes[0] / HH;       // 8192 tokens

  const size_t wElems = (size_t)VV * HH;
  const size_t hElems = (size_t)Ntok * HH;
  const size_t needBytes = (wElems + hElems) * sizeof(unsigned short)
                         + (size_t)2 * Ntok * sizeof(float);

  if (ws_size >= needBytes) {
    unsigned short* Wb  = (unsigned short*)d_ws;
    unsigned short* Hb  = Wb + wElems;
    float*          nll = (float*)(Wb + wElems + hElems);

    cvt_f32_bf16_kernel<<<4096, 256, 0, stream>>>(W,      Wb, (long long)(wElems / 4));
    cvt_f32_bf16_kernel<<<2048, 256, 0, stream>>>(hidden, Hb, (long long)(hElems / 4));

    dim3 ggrid(Ntok / BM, VV / BNF);       // x-fastest reuses W slab in L2
    lmhead_gemm_bf16_kernel<<<ggrid, 256, 0, stream>>>(Hb, Wb, logits);
    lmhead_loss_rows_kernel<<<Ntok, 256, 0, stream>>>(logits, labels, nll, Ntok);
    lmhead_finalize_kernel<<<1, 256, 0, stream>>>(nll, Ntok, out);
  } else {
    float* nll = (float*)d_ws;
    dim3 ggrid(Ntok / BM, VV / 128);
    lmhead_gemm_f32_kernel<<<ggrid, 256, 0, stream>>>(hidden, W, logits);
    lmhead_loss_rows_kernel<<<Ntok, 256, 0, stream>>>(logits, labels, nll, Ntok);
    lmhead_finalize_kernel<<<1, 256, 0, stream>>>(nll, Ntok, out);
  }
}
